// WM_33603824124210
// MI455X (gfx1250) — matrix operand order repaired and checked by reading, not compiled, not run on hardware
//
#include <hip/hip_runtime.h>
#include <hip/hip_bf16.h>
#include <math.h>

// ---------------------------------------------------------------------------
// Sizes (fixed by the reference)
// ---------------------------------------------------------------------------
#define BATCH   4
#define C0      32          // model channels
#define LSEQ    4096        // H*W  (sequence length after the raw reshape)
#define DIN     64          // D_INNER
#define DSTATE  32
#define DTRANK  2
#define NPROJ   66          // DT_RANK + 2*D_STATE
#define NPROJP  80          // padded to multiple of 16 for WMMA
#define MROWS   (BATCH*LSEQ) // 16384 GEMM rows
#define NCHUNK  8
#define LCH     (LSEQ/NCHUNK) // 512

typedef _Float16 v16h __attribute__((ext_vector_type(16)));
typedef _Float16 v8h  __attribute__((ext_vector_type(8)));
typedef float    v8f  __attribute__((ext_vector_type(8)));

__device__ __forceinline__ float wave_sum32(float v) {
    #pragma unroll
    for (int m = 16; m; m >>= 1) v += __shfl_xor(v, m, 32);
    return v;
}
__device__ __forceinline__ float sigmoidf_(float x) { return 1.0f / (1.0f + __expf(-x)); }
__device__ __forceinline__ float softplusf_(float x) {
    return (x > 20.0f) ? x : log1pf(__expf(x));
}

// ---------------------------------------------------------------------------
// WMMA GEMM, K compile-time:  C[M x N] = A[M x K] * Bt^T
//   A  : fp32 row-major (lda), converted to f16 at fragment load (2x v8f loads)
//   Bt : f16 column-major, i.e. stored N x K row-major -> each lane's 16
//        K-values are two 16-byte runs of 8 (2x b128 loads)
//   C  : fp32 row-major (ldc)
//   One 16x16 tile per wave32; K stepped by 32, fully unrolled.
//   Fragment layouts per CDNA5 ISA 7.12.2.
// ---------------------------------------------------------------------------
template <int K>
__global__ __launch_bounds__(256)
void gemm_wmma_kernel(const float* __restrict__ A, const _Float16* __restrict__ Bt,
                      float* __restrict__ C, int M, int N, int lda, int ldc)
{
    const int lane   = threadIdx.x & 31;
    const int wave   = blockIdx.x * (blockDim.x >> 5) + (threadIdx.x >> 5);
    const int ntiles = N >> 4;
    const int mtiles = M >> 4;
    if (wave >= mtiles * ntiles) return;   // wave-uniform; EXEC stays all-1s

    const int m0 = (wave / ntiles) << 4;
    const int n0 = (wave % ntiles) << 4;

    // A-frag: lanes 0-15 row M=lane hold K {0..7,16..23};
    //         lanes 16-31 row M=lane-16 hold K {8..15,24..31}
    const int arow = lane & 15;
    const int akb  = (lane < 16) ? 0 : 8;
    // B-frag: col = lane&15 ; lanes 0-15 K {0..7,16..23}, lanes 16-31 K {8..15,24..31}
    const int bcol = lane & 15;
    const int bkb  = (lane < 16) ? 0 : 8;

    const float*    Arow = A  + (size_t)(m0 + arow) * lda + akb;
    const _Float16* Bcol = Bt + (size_t)(n0 + bcol) * K   + bkb;

    v8f acc = {};
    #pragma unroll
    for (int k0 = 0; k0 < K; k0 += 32) {
        // two aligned 32B fp32 vector loads, cvt to f16
        v8f a0 = *(const v8f*)(Arow + k0);        // K = akb+k0 .. +7
        v8f a1 = *(const v8f*)(Arow + k0 + 16);   // K = akb+k0+16 .. +23
        v16h a;
        #pragma unroll
        for (int j = 0; j < 8; ++j) {
            a[j]     = (_Float16)a0[j];
            a[j + 8] = (_Float16)a1[j];
        }
        // two aligned 16B f16 vector loads (contiguous K run of this column)
        v8h b0 = *(const v8h*)(Bcol + k0);        // K = bkb+k0 .. +7
        v8h b1 = *(const v8h*)(Bcol + k0 + 16);   // K = bkb+k0+16 .. +23
        v16h b = __builtin_shufflevector(b0, b1, 0, 1, 2, 3, 4, 5, 6, 7,
                                                 8, 9, 10, 11, 12, 13, 14, 15);
        acc = __builtin_amdgcn_wmma_f32_16x16x32_f16(
                  false, a, false, b, (short)0, acc, false, false);
    }

    const int crow = (lane < 16) ? 0 : 8;         // D rows 0..7 / 8..15
    float* Cp = C + (size_t)(m0 + crow) * ldc + n0 + (lane & 15);
    #pragma unroll
    for (int r = 0; r < 8; ++r)
        Cp[(size_t)r * ldc] = acc[r];
}

// ---------------------------------------------------------------------------
// Pack weights to f16.  Reference weights are (N_out, K) row-major, which is
// exactly the column-major (N x K) layout the GEMM wants -> plain convert,
// with zero-padding of x_proj rows 66 -> 80.
// ---------------------------------------------------------------------------
__global__ void pack_weights_kernel(const float* __restrict__ in_proj_w,   // (128,32)
                                    const float* __restrict__ x_proj_w,    // (66,64)
                                    const float* __restrict__ out_proj_w,  // (32,64)
                                    _Float16* __restrict__ B1,             // 128x32
                                    _Float16* __restrict__ B2,             // 80x64
                                    _Float16* __restrict__ B3)             // 32x64
{
    int id = blockIdx.x * blockDim.x + threadIdx.x;
    if (id < 128 * 32) {
        B1[id] = (_Float16)in_proj_w[id];
    } else if (id < 128 * 32 + NPROJP * DIN) {
        int t = id - 128 * 32;
        int n = t / DIN;
        B2[t] = (_Float16)((n < NPROJ) ? x_proj_w[t] : 0.0f);
    } else if (id < 128 * 32 + NPROJP * DIN + 32 * DIN) {
        int t = id - (128 * 32 + NPROJP * DIN);
        B3[t] = (_Float16)out_proj_w[t];
    }
}

// ---------------------------------------------------------------------------
// conv1 3x3 (32 -> 64) + ReLU, NCHW direct
// ---------------------------------------------------------------------------
__global__ __launch_bounds__(256)
void conv1_relu_kernel(const float* __restrict__ x, const float* __restrict__ w,
                       const float* __restrict__ bias, float* __restrict__ t)
{
    int id = blockIdx.x * 256 + threadIdx.x;          // (b,oc,h,w) 4*64*64*64
    int ww = id & 63, h = (id >> 6) & 63, oc = (id >> 12) & 63, b = id >> 18;
    float acc = bias[oc];
    const float* xb = x + (size_t)b * C0 * LSEQ;
    const float* wp = w + oc * (C0 * 9);
    for (int ic = 0; ic < C0; ++ic) {
        const float* xc = xb + ic * LSEQ;
        #pragma unroll
        for (int kh = 0; kh < 3; ++kh) {
            int hh2 = h + kh - 1; if ((unsigned)hh2 >= 64u) continue;
            #pragma unroll
            for (int kw = 0; kw < 3; ++kw) {
                int wq = ww + kw - 1; if ((unsigned)wq >= 64u) continue;
                acc += xc[hh2 * 64 + wq] * wp[ic * 9 + kh * 3 + kw];
            }
        }
    }
    t[id] = fmaxf(acc, 0.0f);
}

// ---------------------------------------------------------------------------
// conv2 3x3 (64 -> 32) + residual + LayerNorm (raw-reshape semantics:
// each LN group = 32 consecutive w-pixels of one channel -> lane = w & 31)
// Output xs_ln is bitwise the NCHW layout == GEMM A (16384 x 32) row-major.
// ---------------------------------------------------------------------------
__global__ __launch_bounds__(256)
void conv2_res_ln_kernel(const float* __restrict__ t, const float* __restrict__ w,
                         const float* __restrict__ bias, const float* __restrict__ x,
                         const float* __restrict__ ln_g, const float* __restrict__ ln_b,
                         float* __restrict__ xs)
{
    int id = blockIdx.x * 256 + threadIdx.x;          // (b,c,h,w) 4*32*64*64
    int ww = id & 63, h = (id >> 6) & 63, c = (id >> 12) & 31, b = id >> 17;
    float acc = bias[c];
    const float* tb = t + (size_t)b * 64 * LSEQ;
    const float* wp = w + c * (64 * 9);
    for (int ic = 0; ic < 64; ++ic) {
        const float* tc = tb + ic * LSEQ;
        #pragma unroll
        for (int kh = 0; kh < 3; ++kh) {
            int hh2 = h + kh - 1; if ((unsigned)hh2 >= 64u) continue;
            #pragma unroll
            for (int kw = 0; kw < 3; ++kw) {
                int wq = ww + kw - 1; if ((unsigned)wq >= 64u) continue;
                acc += tc[hh2 * 64 + wq] * wp[ic * 9 + kh * 3 + kw];
            }
        }
    }
    acc += x[id];                                     // residual
    // LayerNorm across wave32 (the 32-element reshape groups)
    float mu = wave_sum32(acc) * (1.0f / 32.0f);
    float dv = (acc - mu) * (acc - mu);
    float var = wave_sum32(dv) * (1.0f / 32.0f);
    int lane = threadIdx.x & 31;
    xs[id] = (acc - mu) * rsqrtf(var + 1e-5f) * ln_g[lane] + ln_b[lane];
}

// ---------------------------------------------------------------------------
// Causal depthwise conv1d (k=4) + SiLU  over xz[:, :64] -> xm_act (B,L,64)
// ---------------------------------------------------------------------------
__global__ __launch_bounds__(256)
void dwconv_silu_kernel(const float* __restrict__ xz, const float* __restrict__ w,
                        const float* __restrict__ bias, float* __restrict__ xm)
{
    int id = blockIdx.x * 256 + threadIdx.x;          // (b,l,d) 4*4096*64
    int d = id & 63, l = (id >> 6) & 4095, b = id >> 18;
    const float* base = xz + (size_t)b * LSEQ * 128 + d;
    float acc = bias[d];
    #pragma unroll
    for (int i = 0; i < 4; ++i) {
        int ls = l - 3 + i;
        if (ls >= 0) acc += w[d * 4 + i] * base[(size_t)ls * 128];
    }
    xm[id] = acc * sigmoidf_(acc);                    // SiLU
}

// ---------------------------------------------------------------------------
// Selective scan, pass 1: per-chunk (prodA, h) partials.
// Wave32: lane = state index n (DSTATE == 32, exact fit).
// dt_proj (K=2) + softplus fused (recomputed in pass 2; arithmetic is free).
// ---------------------------------------------------------------------------
__global__ __launch_bounds__(32)
void scan_part1_kernel(const float* __restrict__ xdbl, const float* __restrict__ xm,
                       const float* __restrict__ dtw, const float* __restrict__ dtb,
                       const float* __restrict__ A_log,
                       float* __restrict__ PA, float* __restrict__ HP)
{
    int n = threadIdx.x;
    int chunk = blockIdx.x & (NCHUNK - 1);
    int d = (blockIdx.x >> 3) & 63;
    int b = blockIdx.x >> 9;
    const float Adn = -__expf(A_log[d * DSTATE + n]);
    const float w0 = dtw[d * 2], w1 = dtw[d * 2 + 1], bd = dtb[d];
    const int l0 = chunk * LCH;
    const float* row = xdbl + ((size_t)b * LSEQ + l0) * NPROJP;
    const float* xmp = xm + ((size_t)b * LSEQ + l0) * DIN + d;
    float pa = 1.0f, h = 0.0f;
    for (int l = 0; l < LCH; ++l) {
        __builtin_prefetch(row + 16 * NPROJP, 0, 1);   // stay ahead of the chain
        float delta = softplusf_(fmaf(row[0], w0, fmaf(row[1], w1, bd)));
        float dA  = __expf(delta * Adn);
        float dBx = delta * row[DTRANK + n] * (*xmp);
        h  = fmaf(dA, h, dBx);
        pa *= dA;
        row += NPROJP; xmp += DIN;
    }
    int idx = blockIdx.x * 32 + n;                    // [b][d][chunk][n]
    PA[idx] = pa; HP[idx] = h;
}

// Pass 2: serial carry scan across the 8 chunks (tiny)
__global__ __launch_bounds__(32)
void scan_carry_kernel(const float* __restrict__ PA, const float* __restrict__ HP,
                       float* __restrict__ CARRY)
{
    int n = threadIdx.x, bd = blockIdx.x;             // 256 blocks
    float c = 0.0f;
    #pragma unroll
    for (int ch = 0; ch < NCHUNK; ++ch) {
        int idx = (bd * NCHUNK + ch) * 32 + n;
        CARRY[idx] = c;
        c = fmaf(PA[idx], c, HP[idx]);
    }
}

// Pass 3: re-scan with carries + y = <h,C> + xm*Dp, gate by SiLU(z) -> yg (B,L,64)
__global__ __launch_bounds__(32)
void scan_part2_kernel(const float* __restrict__ xdbl, const float* __restrict__ xm,
                       const float* __restrict__ xz,
                       const float* __restrict__ dtw, const float* __restrict__ dtb,
                       const float* __restrict__ A_log, const float* __restrict__ Dp,
                       const float* __restrict__ CARRY, float* __restrict__ yg)
{
    int n = threadIdx.x;
    int chunk = blockIdx.x & (NCHUNK - 1);
    int d = (blockIdx.x >> 3) & 63;
    int b = blockIdx.x >> 9;
    const float Adn = -__expf(A_log[d * DSTATE + n]);
    const float w0 = dtw[d * 2], w1 = dtw[d * 2 + 1], bd = dtb[d];
    const float Dpd = Dp[d];
    const int l0 = chunk * LCH;
    const float* row = xdbl + ((size_t)b * LSEQ + l0) * NPROJP;
    const float* xmp = xm + ((size_t)b * LSEQ + l0) * DIN + d;
    const float* zp  = xz + ((size_t)b * LSEQ + l0) * 128 + 64 + d;
    float* yp = yg + ((size_t)b * LSEQ + l0) * DIN + d;
    float h = CARRY[blockIdx.x * 32 + n];
    for (int l = 0; l < LCH; ++l) {
        __builtin_prefetch(row + 16 * NPROJP, 0, 1);
        float delta = softplusf_(fmaf(row[0], w0, fmaf(row[1], w1, bd)));
        float xmv = *xmp;
        float dA  = __expf(delta * Adn);
        float dBx = delta * row[DTRANK + n] * xmv;
        h = fmaf(dA, h, dBx);
        float tv = wave_sum32(h * row[DTRANK + DSTATE + n]);   // sum_n h*C
        if (n == 0) {
            float y = tv + xmv * Dpd;
            float z = *zp;
            *yp = y * (z * sigmoidf_(z));
        }
        row += NPROJP; xmp += DIN; zp += 128; yp += DIN;
    }
}

// ---------------------------------------------------------------------------
// Smooth conv 3x3 (32->32): reads y in (B,L,32) = transpose semantics of the
// reference, writes d_out in NCHW.
// ---------------------------------------------------------------------------
__global__ __launch_bounds__(256)
void smooth_conv_kernel(const float* __restrict__ y, const float* __restrict__ w,
                        const float* __restrict__ bias, float* __restrict__ out)
{
    int id = blockIdx.x * 256 + threadIdx.x;          // (b,oc,h,w) 4*32*64*64
    int ww = id & 63, h = (id >> 6) & 63, oc = (id >> 12) & 31, b = id >> 17;
    float acc = bias[oc];
    const float* wp = w + oc * (C0 * 9);
    const float* yb = y + (size_t)b * LSEQ * C0;
    #pragma unroll
    for (int kh = 0; kh < 3; ++kh) {
        int hh2 = h + kh - 1; if ((unsigned)hh2 >= 64u) continue;
        #pragma unroll
        for (int kw = 0; kw < 3; ++kw) {
            int wq = ww + kw - 1; if ((unsigned)wq >= 64u) continue;
            const float* yp = yb + (size_t)(hh2 * 64 + wq) * C0;
            for (int ic = 0; ic < C0; ++ic)
                acc += yp[ic] * wp[ic * 9 + kh * 3 + kw];
        }
    }
    out[id] = acc;
}

// ---------------------------------------------------------------------------
// Host side
// ---------------------------------------------------------------------------
extern "C" void kernel_launch(void* const* d_in, const int* in_sizes, int n_in,
                              void* d_out, int out_size, void* d_ws, size_t ws_size,
                              hipStream_t stream)
{
    const float* x         = (const float*)d_in[0];
    const float* conv1_w   = (const float*)d_in[1];
    const float* conv1_b   = (const float*)d_in[2];
    const float* conv2_w   = (const float*)d_in[3];
    const float* conv2_b   = (const float*)d_in[4];
    const float* ln_g      = (const float*)d_in[5];
    const float* ln_b      = (const float*)d_in[6];
    const float* in_proj_w = (const float*)d_in[7];
    const float* conv1d_w  = (const float*)d_in[8];
    const float* conv1d_b  = (const float*)d_in[9];
    const float* x_proj_w  = (const float*)d_in[10];
    const float* dt_proj_w = (const float*)d_in[11];
    const float* dt_proj_b = (const float*)d_in[12];
    const float* A_log     = (const float*)d_in[13];
    const float* Dp        = (const float*)d_in[14];
    const float* out_proj_w= (const float*)d_in[15];
    const float* smooth_w  = (const float*)d_in[16];
    const float* smooth_b  = (const float*)d_in[17];
    float* out = (float*)d_out;

    // Workspace layout (256B aligned), total ~31.5 MB
    char* base = (char*)d_ws;
    size_t off = 0;
    auto carve = [&](size_t bytes) { char* p = base + off; off += (bytes + 255) & ~(size_t)255; return p; };
    float*    t_buf  = (float*)   carve((size_t)BATCH * 64 * LSEQ * 4);          // conv1 out
    float*    xs_ln  = (float*)   carve((size_t)MROWS * C0 * 4);                 // LN out / GEMM A
    float*    xz     = (float*)   carve((size_t)MROWS * 128 * 4);                // in_proj out
    float*    xm_act = (float*)   carve((size_t)MROWS * DIN * 4);                // conv1d+SiLU out
    float*    xdbl   = (float*)   carve((size_t)MROWS * NPROJP * 4);             // x_proj out (padded)
    float*    yg     = (float*)   carve((size_t)MROWS * DIN * 4);                // gated scan out
    float*    ypre   = (float*)   carve((size_t)MROWS * C0 * 4);                 // out_proj out
    float*    PA     = (float*)   carve((size_t)BATCH * DIN * NCHUNK * 32 * 4);
    float*    HP     = (float*)   carve((size_t)BATCH * DIN * NCHUNK * 32 * 4);
    float*    CAR    = (float*)   carve((size_t)BATCH * DIN * NCHUNK * 32 * 4);
    _Float16* B1h    = (_Float16*)carve((size_t)128 * 32 * 2);                   // NxK f16
    _Float16* B2h    = (_Float16*)carve((size_t)NPROJP * DIN * 2);               // NxK f16 (padded)
    _Float16* B3h    = (_Float16*)carve((size_t)32 * DIN * 2);                   // NxK f16
    (void)ws_size; (void)in_sizes; (void)n_in; (void)out_size;

    // 0) pack f16 weight matrices for WMMA (already NxK in memory -> convert)
    {
        int total = 128 * 32 + NPROJP * DIN + 32 * DIN;
        pack_weights_kernel<<<(total + 255) / 256, 256, 0, stream>>>(
            in_proj_w, x_proj_w, out_proj_w, B1h, B2h, B3h);
    }
    // 1) conv1 + ReLU
    conv1_relu_kernel<<<(BATCH * 64 * LSEQ) / 256, 256, 0, stream>>>(x, conv1_w, conv1_b, t_buf);
    // 2) conv2 + residual + LayerNorm (fused, wave32 groups)
    conv2_res_ln_kernel<<<(BATCH * C0 * LSEQ) / 256, 256, 0, stream>>>(
        t_buf, conv2_w, conv2_b, x, ln_g, ln_b, xs_ln);
    // 3) in_proj: (16384x32)@(32x128) — K=32, one v_wmma per tile, 8192 waves
    gemm_wmma_kernel<32><<<(MROWS / 16) * (128 / 16) / 8, 256, 0, stream>>>(
        xs_ln, B1h, xz, MROWS, 128, 32, 128);
    // 4) causal depthwise conv1d + SiLU
    dwconv_silu_kernel<<<(MROWS * DIN) / 256, 256, 0, stream>>>(xz, conv1d_w, conv1d_b, xm_act);
    // 5) x_proj: (16384x64)@(64x80) — K=64, two unrolled v_wmma per tile
    gemm_wmma_kernel<64><<<(MROWS / 16) * (NPROJP / 16) / 8, 256, 0, stream>>>(
        xm_act, B2h, xdbl, MROWS, NPROJP, DIN, NPROJP);
    // 6-8) chunked selective scan (lane = state n, wave32 exact fit)
    scan_part1_kernel<<<BATCH * DIN * NCHUNK, 32, 0, stream>>>(
        xdbl, xm_act, dt_proj_w, dt_proj_b, A_log, PA, HP);
    scan_carry_kernel<<<BATCH * DIN, 32, 0, stream>>>(PA, HP, CAR);
    scan_part2_kernel<<<BATCH * DIN * NCHUNK, 32, 0, stream>>>(
        xdbl, xm_act, xz, dt_proj_w, dt_proj_b, A_log, Dp, CAR, yg);
    // 9) out_proj: (16384x64)@(64x32) — K=64
    gemm_wmma_kernel<64><<<(MROWS / 16) * (C0 / 16) / 8, 256, 0, stream>>>(
        yg, B3h, ypre, MROWS, C0, DIN, C0);
    // 10) smooth conv (reads transpose-semantics (B,L,32), writes NCHW)
    smooth_conv_kernel<<<(BATCH * C0 * LSEQ) / 256, 256, 0, stream>>>(
        ypre, smooth_w, smooth_b, out);
}
